// MultiheadAttention_58720792871541
// MI455X (gfx1250) — compile-verified, matrix operand order checked by reading, not run on hardware
//
#include <hip/hip_runtime.h>
#include <hip/hip_bf16.h>

typedef __attribute__((ext_vector_type(16))) _Float16 v16h;
typedef __attribute__((ext_vector_type(8)))  _Float16 v8h;
typedef __attribute__((ext_vector_type(8)))  float    v8f;

#define B_DIM 8
#define S_DIM 1024
#define E_DIM 1024
#define H_DIM 16
#define DH_DIM 64
#define SC_STRIDE 1028   // 1024 + 4 floats pad: removes LDS bank conflicts

// ---------------------------------------------------------------------------
// Fragment loader for 16-bit WMMA operands (A: lane=m, B: lane=n; both read
// contiguous-K rows). Per ISA 7.12.2: lane L holds K-chunks
// [(L/16)*8, +8) and [16+(L/16)*8, +8)  -> two 16-byte loads.
// ---------------------------------------------------------------------------
__device__ __forceinline__ v16h load_frag_k16(const _Float16* base, int strideHalves, int lane) {
    const int row = lane & 15;
    const int c0  = (lane >> 4) << 3;
    const _Float16* p = base + (size_t)row * strideHalves + c0;
    union { v16h v; v8h h[2]; } u;
    u.h[0] = *(const v8h*)(p);
    u.h[1] = *(const v8h*)(p + 16);
    return u.v;
}

// ---------------------------------------------------------------------------
// f32 -> f16 conversion
// ---------------------------------------------------------------------------
__global__ void mha_cvt_f16_kernel(const float* __restrict__ src, _Float16* __restrict__ dst, int n) {
    int i = blockIdx.x * blockDim.x + threadIdx.x;
    if (i < n) dst[i] = (_Float16)src[i];
}

// ---------------------------------------------------------------------------
// Projection GEMM: Y[m,n] = sum_k X[m,k] * W[n,k] + bias[n]
//   X: [8192,1024] f16, W: [1024,1024] f16 ([N,K] row-major)
//   mode 0: store f16 to [B,H,S,Dh]   (Q / K heads)
//   mode 1: store f16 to [B,H,Dh,S]   (V transposed)
//   mode 2: store f32 to [M,1024]     (final output)
// One wave per 16(M) x 128(N) tile; 4 waves per block.
// ---------------------------------------------------------------------------
__global__ __launch_bounds__(128) void mha_proj_gemm_kernel(
    const _Float16* __restrict__ X, const _Float16* __restrict__ W,
    const float* __restrict__ bias,
    _Float16* __restrict__ out16, float* __restrict__ out32, int mode)
{
    const int lane  = threadIdx.x & 31;
    const int wid   = blockIdx.x * 4 + (threadIdx.x >> 5);
    const int mtile = wid >> 3;   // 0..511
    const int nblk  = wid & 7;    // 0..7 -> 128 columns

    const _Float16* xrow = X + (size_t)mtile * 16 * E_DIM;

    v8f acc[8] = {};
    for (int k0 = 0; k0 < E_DIM; k0 += 32) {
        v16h aF = load_frag_k16(xrow + k0, E_DIM, lane);
#pragma unroll
        for (int t = 0; t < 8; ++t) {
            const _Float16* wrow = W + (size_t)(nblk * 128 + t * 16) * E_DIM + k0;
            v16h bF = load_frag_k16(wrow, E_DIM, lane);
            acc[t] = __builtin_amdgcn_wmma_f32_16x16x32_f16(
                false, aF, false, bF, (short)0, acc[t], false, false);
        }
    }

    const int rowBase = mtile * 16 + ((lane >> 4) << 3);
#pragma unroll
    for (int t = 0; t < 8; ++t) {
        const int n  = nblk * 128 + t * 16 + (lane & 15);
        const float bv = bias[n];
#pragma unroll
        for (int v = 0; v < 8; ++v) {
            const int m = rowBase + v;
            const float y = acc[t][v] + bv;
            if (mode == 2) {
                out32[(size_t)m * E_DIM + n] = y;
            } else {
                const int b = m >> 10, s = m & 1023;
                const int h = n >> 6,  d = n & 63;
                if (mode == 0)
                    out16[((size_t)(b * H_DIM + h) * S_DIM + s) * DH_DIM + d] = (_Float16)y;
                else
                    out16[((size_t)(b * H_DIM + h) * DH_DIM + d) * S_DIM + s] = (_Float16)y;
            }
        }
    }
}

// ---------------------------------------------------------------------------
// Attention core: per (b,h,64-row q-block).
//   scores = Q K^T / 8  -> LDS (64 x 1024 fp32, 257KB of the 320KB WGP LDS)
//   softmax rows in LDS -> write attn fp32 to global
//   ctx = attn @ V      -> f16 ctx buffer [B,S,E]
// 4 waves x 16-row WMMA stripes; each wave's stripe is private (no barriers).
// ---------------------------------------------------------------------------
__global__ __launch_bounds__(128) void mha_attn_kernel(
    const _Float16* __restrict__ qh,   // [B,H,S,Dh]
    const _Float16* __restrict__ kh,   // [B,H,S,Dh]
    const _Float16* __restrict__ vt,   // [B,H,Dh,S]
    float* __restrict__ attn_out,      // [B,H,S,S]
    _Float16* __restrict__ ctx16)      // [B,S,E]
{
    extern __shared__ float sc[];
    const int lane = threadIdx.x & 31;
    const int wave = threadIdx.x >> 5;
    const int bh   = blockIdx.x >> 4;  // 0..127 = b*16+h
    const int qblk = blockIdx.x & 15;  // 0..15
    const int b    = bh >> 4, h = bh & 15;
    const int qrow0 = qblk * 64 + wave * 16;

    // ---- scores: Q(16xDh) x K^T, Dh=64 -> two K=32 WMMA steps per tile ----
    const _Float16* qbase = qh + ((size_t)bh * S_DIM + qrow0) * DH_DIM;
    const v16h aF0 = load_frag_k16(qbase,      DH_DIM, lane);
    const v16h aF1 = load_frag_k16(qbase + 32, DH_DIM, lane);

    float* myrows = sc + (size_t)wave * 16 * SC_STRIDE;

    for (int ct = 0; ct < 64; ++ct) {
        const _Float16* kbase = kh + ((size_t)bh * S_DIM + ct * 16) * DH_DIM;
        v16h bF0 = load_frag_k16(kbase,      DH_DIM, lane);
        v16h bF1 = load_frag_k16(kbase + 32, DH_DIM, lane);
        v8f c = {};
        c = __builtin_amdgcn_wmma_f32_16x16x32_f16(false, aF0, false, bF0, (short)0, c, false, false);
        c = __builtin_amdgcn_wmma_f32_16x16x32_f16(false, aF1, false, bF1, (short)0, c, false, false);
        const int col = ct * 16 + (lane & 15);
        const int r0  = (lane >> 4) << 3;
#pragma unroll
        for (int v = 0; v < 8; ++v)
            myrows[(r0 + v) * SC_STRIDE + col] = c[v] * 0.125f;  // 1/sqrt(64)
    }

    // ---- softmax over each of this wave's 16 rows (wave32 reductions) ----
    for (int r = 0; r < 16; ++r) {
        float* row = myrows + r * SC_STRIDE;
        float m = -3.402823466e38f;
        for (int i = lane; i < S_DIM; i += 32) m = fmaxf(m, row[i]);
#pragma unroll
        for (int off = 16; off > 0; off >>= 1) m = fmaxf(m, __shfl_xor(m, off, 32));
        float s = 0.f;
        for (int i = lane; i < S_DIM; i += 32) {
            float e = __expf(row[i] - m);
            row[i] = e;
            s += e;
        }
#pragma unroll
        for (int off = 16; off > 0; off >>= 1) s += __shfl_xor(s, off, 32);
        const float inv = 1.0f / s;
        for (int i = lane; i < S_DIM; i += 32) row[i] *= inv;
    }

    // ---- stream attn fp32 to global (coalesced) ----
    {
        float* dst = attn_out + ((size_t)bh * S_DIM + qrow0) * S_DIM;
        for (int r = 0; r < 16; ++r)
            for (int i = lane; i < S_DIM; i += 32)
                dst[(size_t)r * S_DIM + i] = myrows[r * SC_STRIDE + i];
    }

    // ---- ctx = attn(16x1024) @ V(1024x64): A from LDS (f32->f16), B = Vt ----
    v8f acc[4] = {};
    const _Float16* vbase = vt + (size_t)bh * DH_DIM * S_DIM;
    for (int k0 = 0; k0 < S_DIM; k0 += 32) {
        const int arow = lane & 15;
        const int c0   = k0 + ((lane >> 4) << 3);
        const float* ap = myrows + (size_t)arow * SC_STRIDE + c0;
        float4 f0 = *(const float4*)(ap);
        float4 f1 = *(const float4*)(ap + 4);
        float4 f2 = *(const float4*)(ap + 16);
        float4 f3 = *(const float4*)(ap + 20);
        union { v16h v; _Float16 e[16]; } a;
        a.e[0]=(_Float16)f0.x;  a.e[1]=(_Float16)f0.y;  a.e[2]=(_Float16)f0.z;  a.e[3]=(_Float16)f0.w;
        a.e[4]=(_Float16)f1.x;  a.e[5]=(_Float16)f1.y;  a.e[6]=(_Float16)f1.z;  a.e[7]=(_Float16)f1.w;
        a.e[8]=(_Float16)f2.x;  a.e[9]=(_Float16)f2.y;  a.e[10]=(_Float16)f2.z; a.e[11]=(_Float16)f2.w;
        a.e[12]=(_Float16)f3.x; a.e[13]=(_Float16)f3.y; a.e[14]=(_Float16)f3.z; a.e[15]=(_Float16)f3.w;
#pragma unroll
        for (int t = 0; t < 4; ++t) {
            v16h bF = load_frag_k16(vbase + (size_t)(t * 16) * S_DIM + k0, S_DIM, lane);
            acc[t] = __builtin_amdgcn_wmma_f32_16x16x32_f16(
                false, a.v, false, bF, (short)0, acc[t], false, false);
        }
    }

    // store ctx f16 to [b, s, h*64+d]
    const int r0 = (lane >> 4) << 3;
#pragma unroll
    for (int t = 0; t < 4; ++t) {
        const int d = t * 16 + (lane & 15);
#pragma unroll
        for (int v = 0; v < 8; ++v) {
            const int srow = qrow0 + r0 + v;
            ctx16[((size_t)b * S_DIM + srow) * E_DIM + h * DH_DIM + d] = (_Float16)acc[t][v];
        }
    }
}

// ---------------------------------------------------------------------------
// Host launcher
// ---------------------------------------------------------------------------
extern "C" void kernel_launch(void* const* d_in, const int* in_sizes, int n_in,
                              void* d_out, int out_size, void* d_ws, size_t ws_size,
                              hipStream_t stream) {
    (void)in_sizes; (void)n_in; (void)out_size; (void)ws_size;

    const float* query = (const float*)d_in[0];
    const float* key   = (const float*)d_in[1];
    const float* value = (const float*)d_in[2];
    const float* Wq = (const float*)d_in[3];  const float* bq = (const float*)d_in[4];
    const float* Wk = (const float*)d_in[5];  const float* bk = (const float*)d_in[6];
    const float* Wv = (const float*)d_in[7];  const float* bv = (const float*)d_in[8];
    const float* Wo = (const float*)d_in[9];  const float* bo = (const float*)d_in[10];

    const size_t NX = (size_t)B_DIM * S_DIM * E_DIM;   // 8M elements
    const size_t NW = (size_t)E_DIM * E_DIM;           // 1M elements

    _Float16* ws = (_Float16*)d_ws;
    size_t off = 0;
    _Float16* x16q = ws + off; off += NX;
    _Float16* x16k = ws + off; off += NX;
    _Float16* x16v = ws + off; off += NX;
    _Float16* wq16 = ws + off; off += NW;
    _Float16* wk16 = ws + off; off += NW;
    _Float16* wv16 = ws + off; off += NW;
    _Float16* wo16 = ws + off; off += NW;
    _Float16* qh   = ws + off; off += NX;
    _Float16* kh   = ws + off; off += NX;
    _Float16* vt   = ws + off; off += NX;
    _Float16* ctx  = x16q;  // query-f16 is dead by the time ctx is produced

    float* out_f32  = (float*)d_out;
    float* attn_f32 = (float*)d_out + NX;   // (out, attn) concatenated

    // 1) convert inputs + weights to f16
    {
        const int T = 256;
        mha_cvt_f16_kernel<<<(int)((NX + T - 1) / T), T, 0, stream>>>(query, x16q, (int)NX);
        mha_cvt_f16_kernel<<<(int)((NX + T - 1) / T), T, 0, stream>>>(key,   x16k, (int)NX);
        mha_cvt_f16_kernel<<<(int)((NX + T - 1) / T), T, 0, stream>>>(value, x16v, (int)NX);
        mha_cvt_f16_kernel<<<(int)((NW + T - 1) / T), T, 0, stream>>>(Wq, wq16, (int)NW);
        mha_cvt_f16_kernel<<<(int)((NW + T - 1) / T), T, 0, stream>>>(Wk, wk16, (int)NW);
        mha_cvt_f16_kernel<<<(int)((NW + T - 1) / T), T, 0, stream>>>(Wv, wv16, (int)NW);
        mha_cvt_f16_kernel<<<(int)((NW + T - 1) / T), T, 0, stream>>>(Wo, wo16, (int)NW);
    }

    // 2) QKV projections (V stored head-transposed)
    mha_proj_gemm_kernel<<<1024, 128, 0, stream>>>(x16q, wq16, bq, qh, nullptr, 0);
    mha_proj_gemm_kernel<<<1024, 128, 0, stream>>>(x16k, wk16, bk, kh, nullptr, 0);
    mha_proj_gemm_kernel<<<1024, 128, 0, stream>>>(x16v, wv16, bv, vt, nullptr, 1);

    // 3) attention core (scores+softmax+attn-write+ctx), 257KB LDS per WG
    {
        const size_t ldsBytes = (size_t)64 * SC_STRIDE * sizeof(float);
        hipFuncSetAttribute((const void*)mha_attn_kernel,
                            hipFuncAttributeMaxDynamicSharedMemorySize, (int)ldsBytes);
        mha_attn_kernel<<<B_DIM * H_DIM * (S_DIM / 64), 128, ldsBytes, stream>>>(
            qh, kh, vt, attn_f32, ctx);
    }

    // 4) output projection (fp32 result)
    mha_proj_gemm_kernel<<<1024, 128, 0, stream>>>(ctx, wo16, bo, nullptr, out_f32, 2);
}
